// RNN_64433099374927
// MI455X (gfx1250) — matrix operand order repaired and checked by reading, not compiled, not run on hardware
//
#include <hip/hip_runtime.h>
#include <hip/hip_bf16.h>

// ---------------------------------------------------------------------------
// Problem constants (from reference): B=32, T=512, D_IN=512, H=1024, D_OUT=128
// ---------------------------------------------------------------------------
#define BB      32
#define TT      512
#define DIN     512
#define HH      1024
#define DOUT    128
#define ALPHA   0.1f
#define OMALPHA 0.9f

#define REC_NWG 32          // workgroups in the persistent recurrent kernel
#define COLS_WG 32          // H columns owned per workgroup (32*32 = 1024)
#define LDSK    (HH + 8)    // padded column length (bf16 elems) -> conflict-free

typedef __attribute__((ext_vector_type(16))) __bf16 v16bf;
typedef __attribute__((ext_vector_type(8)))  float  v8f;
typedef __attribute__((ext_vector_type(4)))  unsigned v4u;
typedef __attribute__((ext_vector_type(8)))  unsigned v8u;

__device__ __forceinline__ v8f wmma_bf16(v16bf a, v16bf b, v8f c) {
    // D = A(16x32 bf16) x B(32x16 bf16) + C(16x16 f32)
    return __builtin_amdgcn_wmma_f32_16x16x32_bf16(
        /*neg_a=*/false, a, /*neg_b=*/false, b,
        /*c_mod=*/(short)0, c, /*reuse_a=*/false, /*reuse_b=*/false);
}

// A-fragment from a bf16 row (row-major [rows][K]); ISA 16-bit A layout:
// lanes 0-15 hold K = {k0..k0+7, k0+16..k0+23}; lanes 16-31 the +8 chunks.
__device__ __forceinline__ v16bf a_frag_bf16(const __bf16* rowp, int k0, int ksel) {
    union { v16bf v; uint4 u[2]; } r;
    r.u[0] = *(const uint4*)(rowp + k0 + 8 * ksel);
    r.u[1] = *(const uint4*)(rowp + k0 + 16 + 8 * ksel);
    return r.v;
}

// A-fragment gathered from f32 memory with inline f32->bf16 conversion.
__device__ __forceinline__ v16bf a_frag_f32(const float* rowp, int k0, int ksel) {
    const float4 f0 = *(const float4*)(rowp + k0 + 8 * ksel);
    const float4 f1 = *(const float4*)(rowp + k0 + 8 * ksel + 4);
    const float4 g0 = *(const float4*)(rowp + k0 + 16 + 8 * ksel);
    const float4 g1 = *(const float4*)(rowp + k0 + 16 + 8 * ksel + 4);
    v16bf a;
    a[0]  = (__bf16)f0.x; a[1]  = (__bf16)f0.y; a[2]  = (__bf16)f0.z; a[3]  = (__bf16)f0.w;
    a[4]  = (__bf16)f1.x; a[5]  = (__bf16)f1.y; a[6]  = (__bf16)f1.z; a[7]  = (__bf16)f1.w;
    a[8]  = (__bf16)g0.x; a[9]  = (__bf16)g0.y; a[10] = (__bf16)g0.z; a[11] = (__bf16)g0.w;
    a[12] = (__bf16)g1.x; a[13] = (__bf16)g1.y; a[14] = (__bf16)g1.z; a[15] = (__bf16)g1.w;
    return a;
}

// B-fragment: weights stored column-major ([N][K] contiguous K). 16-bit B
// layout matching A: lane n holds K = {k0+8*(lane/16)..+7, k0+16+8*(lane/16)..+7}.
__device__ __forceinline__ v16bf b_frag(const __bf16* colp, int k0, int ksel) {
    union { v16bf v; uint4 u[2]; } r;
    const __bf16* q = colp + k0 + 8 * ksel;
    r.u[0] = *(const uint4*)(q);
    r.u[1] = *(const uint4*)(q + 16);
    return r.v;
}

// ---------------------------------------------------------------------------
// Tensor Data Mover: async 2D bf16 tile load (global -> LDS). D# per ISA §8:
//   group0: count | lds_addr | global_addr[56:0] | type=2
//   group1: data_size=2B | tensor_dim0/1 | tile_dim0/1 | tensor_dim0_stride
// Issued from one wave; completion via s_wait_tensorcnt (TENSORcnt).
// ---------------------------------------------------------------------------
__device__ __forceinline__ void tdm_load_tile_b16(const void* gsrc, unsigned lds_byte,
                                                  unsigned tile_x, unsigned tile_y,
                                                  unsigned stride_elems) {
    unsigned long long ga = (unsigned long long)gsrc;
    v4u g0;
    g0[0] = 1u;                                            // count = 1 (valid D#)
    g0[1] = lds_byte;                                      // lds_addr (bytes)
    g0[2] = (unsigned)ga;                                  // global_addr[31:0]
    g0[3] = (unsigned)((ga >> 32) & 0x01FFFFFFull)         // global_addr[56:32]
            | (2u << 30);                                  // type = 2 ("image")
    v8u g1;
    g1[0] = (1u << 16);                                    // data_size = 1 -> 2 bytes
    g1[1] = (tile_x & 0xFFFFu) << 16;                      // tensor_dim0[15:0]
    g1[2] = ((tile_x >> 16) & 0xFFFFu)                     // tensor_dim0[31:16]
            | ((tile_y & 0xFFFFu) << 16);                  // tensor_dim1[15:0]
    g1[3] = ((tile_y >> 16) & 0xFFFFu)                     // tensor_dim1[31:16]
            | (tile_x << 16);                              // tile_dim0
    g1[4] = (tile_y & 0xFFFFu);                            // tile_dim1 (tile_dim2=0)
    g1[5] = stride_elems;                                  // tensor_dim0_stride[31:0]
    g1[6] = 0u;                                            // stride hi / dim1_stride lo
    g1[7] = 0u;
    asm volatile("tensor_load_to_lds %0, %1" :: "s"(g0), "s"(g1) : "memory");
}

// ---------------------------------------------------------------------------
// Device-wide split barrier (persistent-kernel grid sync)
// bar[0] = arrival counter, bar[1] = generation
// ---------------------------------------------------------------------------
__device__ __forceinline__ void grid_sync(unsigned* bar) {
    __syncthreads();
    if (threadIdx.x == 0) {
        __threadfence();
        unsigned gen  = __hip_atomic_load(bar + 1, __ATOMIC_RELAXED, __HIP_MEMORY_SCOPE_AGENT);
        unsigned prev = __hip_atomic_fetch_add(bar, 1u, __ATOMIC_ACQ_REL, __HIP_MEMORY_SCOPE_AGENT);
        if (prev == (unsigned)(REC_NWG - 1)) {
            __hip_atomic_store(bar, 0u, __ATOMIC_RELAXED, __HIP_MEMORY_SCOPE_AGENT);
            __hip_atomic_fetch_add(bar + 1, 1u, __ATOMIC_ACQ_REL, __HIP_MEMORY_SCOPE_AGENT);
        } else {
            while (__hip_atomic_load(bar + 1, __ATOMIC_ACQUIRE, __HIP_MEMORY_SCOPE_AGENT) == gen) {
                __builtin_amdgcn_s_sleep(8);
            }
        }
        __threadfence();
    }
    __syncthreads();
}

// ---------------------------------------------------------------------------
// 1) Transpose + f32->bf16 convert: src [K][N] row-major -> dst [N][K]
// ---------------------------------------------------------------------------
__global__ void k_cvt_transpose(const float* __restrict__ src, __bf16* __restrict__ dst,
                                int K, int N) {
    size_t total = (size_t)K * N;
    for (size_t i = (size_t)blockIdx.x * blockDim.x + threadIdx.x; i < total;
         i += (size_t)gridDim.x * blockDim.x) {
        size_t k = i / N;
        size_t n = i - k * N;
        dst[n * (size_t)K + k] = (__bf16)src[i];
    }
}

// ---------------------------------------------------------------------------
// 2) Zero the fr0/fr1 double buffers and the barrier words
// ---------------------------------------------------------------------------
__global__ void k_init(__bf16* __restrict__ fr0, __bf16* __restrict__ fr1,
                       unsigned* __restrict__ bar) {
    size_t i = (size_t)blockIdx.x * blockDim.x + threadIdx.x;
    size_t n = (size_t)2 * BB * HH;
    if (i < n) { fr0[i] = (__bf16)0.f; fr1[i] = (__bf16)0.f; }
    if (i < 8) bar[i] = 0u;
}

// ---------------------------------------------------------------------------
// 3) P0[t][b][h] = ALPHA * (x[b,t,:] @ W_in0[:,h] + b0[h])   (bf16 out)
//    One 16x16 tile per wave; two interleaved WMMA accumulator chains.
// ---------------------------------------------------------------------------
__global__ void k_p0_gemm(const float* __restrict__ x, const __bf16* __restrict__ Win0c,
                          const float* __restrict__ b0, __bf16* __restrict__ P0) {
    const int wave = threadIdx.x >> 5;
    const int lane = threadIdx.x & 31;
    const int tile = blockIdx.x * (blockDim.x >> 5) + wave;   // 65536 tiles
    const int mt = tile >> 6;                                 // 1024 m-tiles
    const int nt = tile & 63;                                 // 64 n-tiles
    const int arow = lane & 15, ksel = lane >> 4;

    const int rglob = mt * 16 + arow;        // row in (t,b) order
    const int t = rglob >> 5, b = rglob & 31;
    const float* xrow = x + ((size_t)b * TT + t) * DIN;             // x[b][t][:]
    const __bf16* bcol = Win0c + (size_t)(nt * 16 + (lane & 15)) * DIN;

    v8f accA = {0.f, 0.f, 0.f, 0.f, 0.f, 0.f, 0.f, 0.f};
    v8f accB = {0.f, 0.f, 0.f, 0.f, 0.f, 0.f, 0.f, 0.f};
#pragma unroll 4
    for (int k0 = 0; k0 < DIN / 2; k0 += 32) {
        accA = wmma_bf16(a_frag_f32(xrow, k0, ksel), b_frag(bcol, k0, ksel), accA);
        accB = wmma_bf16(a_frag_f32(xrow, k0 + 256, ksel), b_frag(bcol, k0 + 256, ksel), accB);
    }
    v8f acc = accA + accB;

    const int colg = nt * 16 + (lane & 15);
    const float bias = b0[colg];
#pragma unroll
    for (int r = 0; r < 8; ++r) {
        int row = mt * 16 + r + 8 * ksel;     // C layout: M = r + 8*(lane/16)
        int tr = row >> 5, br = row & 31;
        P0[((size_t)tr * BB + br) * HH + colg] = (__bf16)(ALPHA * (acc[r] + bias));
    }
}

// ---------------------------------------------------------------------------
// 4) Persistent recurrent kernel: 32 WGs x 4 waves, LDS-resident weights,
//    TDM double-ahead staging of the per-step P0 tile into LDS.
// ---------------------------------------------------------------------------
__global__ void k_recurrent(const __bf16* __restrict__ P0,
                            const __bf16* __restrict__ Wrec0c,
                            const __bf16* __restrict__ Win1c,
                            const __bf16* __restrict__ Wrec1c,
                            const float* __restrict__ b1,
                            float* __restrict__ hs0, float* __restrict__ hs1,
                            __bf16* __restrict__ fr0buf, __bf16* __restrict__ fr1buf,
                            unsigned* __restrict__ bar) {
    extern __shared__ __bf16 smem[];   // 3*COLS_WG*LDSK weights + 32x32 P0 tile
    const int col0 = blockIdx.x * COLS_WG;

    // Stage this WG's column slices of the three recurrent weights into LDS.
    {
        const __bf16* wsrc[3] = {Wrec0c, Win1c, Wrec1c};
        for (int idx = threadIdx.x; idx < 3 * COLS_WG * (HH / 8); idx += blockDim.x) {
            int w   = idx / (COLS_WG * (HH / 8));
            int rem = idx - w * COLS_WG * (HH / 8);
            int c   = rem >> 7;            // / (HH/8)
            int kc  = rem & 127;
            uint4 v = *(const uint4*)(wsrc[w] + (size_t)(col0 + c) * HH + kc * 8);
            *(uint4*)(&smem[(size_t)w * COLS_WG * LDSK + (size_t)c * LDSK + kc * 8]) = v;
        }
    }
    __bf16* p0tile = smem + (size_t)3 * COLS_WG * LDSK;                 // 32x32 bf16
    const unsigned ldsP0 = (unsigned)__builtin_amdgcn_groupstaticsize()
                         + 3u * COLS_WG * LDSK * 2u;                    // byte offset
    __syncthreads();

    const int wave = threadIdx.x >> 5;     // 0..3
    const int lane = threadIdx.x & 31;
    const int m16   = (wave & 1) * 16;     // batch-row half
    const int ntile = wave >> 1;           // 0..1 within the 32 owned columns
    const int arow = lane & 15, ksel = lane >> 4;
    const int ncol_l = ntile * 16 + (lane & 15);
    const int colg   = col0 + ncol_l;

    const __bf16* w0c = &smem[(size_t)0 * COLS_WG * LDSK + (size_t)ncol_l * LDSK];
    const __bf16* w1c = &smem[(size_t)1 * COLS_WG * LDSK + (size_t)ncol_l * LDSK];
    const __bf16* w2c = &smem[(size_t)2 * COLS_WG * LDSK + (size_t)ncol_l * LDSK];
    const float ab1 = ALPHA * b1[colg];

    // Prime the TDM pipeline with the step-0 P0 tile (rows=batch, cols=owned H).
    if (wave == 0)
        tdm_load_tile_b16(P0 + col0, ldsP0, COLS_WG, BB, HH);

    v8f v0 = {0.f, 0.f, 0.f, 0.f, 0.f, 0.f, 0.f, 0.f};
    v8f v1 = {0.f, 0.f, 0.f, 0.f, 0.f, 0.f, 0.f, 0.f};

    for (int t = 0; t < TT; ++t) {
        const __bf16* fr0_rd = fr0buf + (size_t)(t & 1) * (BB * HH);
        __bf16*       fr0_wr = fr0buf + (size_t)((t + 1) & 1) * (BB * HH);
        const __bf16* fr1_rd = fr1buf + (size_t)(t & 1) * (BB * HH);
        __bf16*       fr1_wr = fr1buf + (size_t)((t + 1) & 1) * (BB * HH);

        // ------- layer 0: v0 = .9*v0 + P0[t] + .1*(fr0(t-1) @ Wrec0) --------
        // Two interleaved accumulator chains over the K halves for WMMA ILP.
        v8f accA = {0.f, 0.f, 0.f, 0.f, 0.f, 0.f, 0.f, 0.f};
        v8f accB = {0.f, 0.f, 0.f, 0.f, 0.f, 0.f, 0.f, 0.f};
        const __bf16* a0 = fr0_rd + (size_t)(m16 + arow) * HH;
#pragma unroll 4
        for (int k0 = 0; k0 < HH / 2; k0 += 32) {
            accA = wmma_bf16(a_frag_bf16(a0, k0, ksel), b_frag(w0c, k0, ksel), accA);
            accB = wmma_bf16(a_frag_bf16(a0, k0 + 512, ksel), b_frag(w0c, k0 + 512, ksel), accB);
        }
        v8f acc = accA + accB;

        // TDM tile for this step is in flight since last iteration: drain it.
        if (wave == 0) __builtin_amdgcn_s_wait_tensorcnt(0);
        __syncthreads();

#pragma unroll
        for (int r = 0; r < 8; ++r) {
            int brow = m16 + r + 8 * ksel;
            float p0v = (float)p0tile[(size_t)brow * COLS_WG + ncol_l];
            v0[r] = OMALPHA * v0[r] + p0v + ALPHA * acc[r];
            float fr = fmaxf(v0[r], 0.f);
            hs0[((size_t)brow * TT + t) * HH + colg] = fr;       // hs0[b][t][h]
            fr0_wr[(size_t)brow * HH + colg] = (__bf16)fr;
        }
        grid_sync(bar);   // all WGs: fr0(t) visible; all waves done with p0tile

        // Issue next step's P0 tile DMA; overlaps phase B + next phase A.
        if (t + 1 < TT && wave == 0)
            tdm_load_tile_b16(P0 + (size_t)(t + 1) * (BB * HH) + col0, ldsP0,
                              COLS_WG, BB, HH);
        if (t + 2 < TT)   // keep HBM->L2 two steps ahead
            __builtin_prefetch(P0 + (size_t)(t + 2) * (BB * HH) +
                               (size_t)(m16 + arow) * HH + col0, 0, 1);

        // --- layer 1: v1 = .9*v1 + .1*(fr0(t)@Win1 + fr1(t-1)@Wrec1 + b1) ---
        v8f accC = {0.f, 0.f, 0.f, 0.f, 0.f, 0.f, 0.f, 0.f};
        v8f accD = {0.f, 0.f, 0.f, 0.f, 0.f, 0.f, 0.f, 0.f};
        const __bf16* a1 = fr0_wr + (size_t)(m16 + arow) * HH;   // fr0(t)
        const __bf16* a2 = fr1_rd + (size_t)(m16 + arow) * HH;   // fr1(t-1)
#pragma unroll 4
        for (int k0 = 0; k0 < HH; k0 += 32) {
            accC = wmma_bf16(a_frag_bf16(a1, k0, ksel), b_frag(w1c, k0, ksel), accC);
            accD = wmma_bf16(a_frag_bf16(a2, k0, ksel), b_frag(w2c, k0, ksel), accD);
        }
        v8f acc1 = accC + accD;

#pragma unroll
        for (int r = 0; r < 8; ++r) {
            int brow = m16 + r + 8 * ksel;
            v1[r] = OMALPHA * v1[r] + ALPHA * acc1[r] + ab1;
            float fr = fmaxf(v1[r], 0.f);
            hs1[((size_t)brow * TT + t) * HH + colg] = fr;       // hs1[b][t][h]
            fr1_wr[(size_t)brow * HH + colg] = (__bf16)fr;
        }
        grid_sync(bar);
    }
}

// ---------------------------------------------------------------------------
// 5) Readout: out[bt][:] = hs1[bt][:] @ W_out + b_out  (A converted inline)
// ---------------------------------------------------------------------------
__global__ void k_readout(const float* __restrict__ hs1, const __bf16* __restrict__ Woutc,
                          const float* __restrict__ bout, float* __restrict__ out) {
    const int wave = threadIdx.x >> 5;
    const int lane = threadIdx.x & 31;
    const int tile = blockIdx.x * (blockDim.x >> 5) + wave;   // 8192 tiles
    const int mt = tile >> 3;                                 // 1024 m-tiles
    const int nt = tile & 7;                                  // 8 n-tiles
    const int arow = lane & 15, ksel = lane >> 4;

    const float* arp = hs1 + (size_t)(mt * 16 + arow) * HH;
    const __bf16* bcol = Woutc + (size_t)(nt * 16 + (lane & 15)) * HH;

    v8f accA = {0.f, 0.f, 0.f, 0.f, 0.f, 0.f, 0.f, 0.f};
    v8f accB = {0.f, 0.f, 0.f, 0.f, 0.f, 0.f, 0.f, 0.f};
#pragma unroll 4
    for (int k0 = 0; k0 < HH / 2; k0 += 32) {
        accA = wmma_bf16(a_frag_f32(arp, k0, ksel), b_frag(bcol, k0, ksel), accA);
        accB = wmma_bf16(a_frag_f32(arp, k0 + 512, ksel), b_frag(bcol, k0 + 512, ksel), accB);
    }
    v8f acc = accA + accB;

    const int colg = nt * 16 + (lane & 15);
    const float bv = bout[colg];
#pragma unroll
    for (int r = 0; r < 8; ++r) {
        int row = mt * 16 + r + 8 * ksel;
        out[(size_t)row * DOUT + colg] = acc[r] + bv;
    }
}

// ---------------------------------------------------------------------------
extern "C" void kernel_launch(void* const* d_in, const int* in_sizes, int n_in,
                              void* d_out, int out_size, void* d_ws, size_t ws_size,
                              hipStream_t stream) {
    const float* x     = (const float*)d_in[0];
    const float* Win0  = (const float*)d_in[1];
    const float* Wrec0 = (const float*)d_in[2];
    const float* b0    = (const float*)d_in[3];
    const float* Win1  = (const float*)d_in[4];
    const float* Wrec1 = (const float*)d_in[5];
    const float* b1    = (const float*)d_in[6];
    const float* Wout  = (const float*)d_in[7];
    const float* bout  = (const float*)d_in[8];

    float* out = (float*)d_out;
    float* hs0 = out + (size_t)BB * TT * DOUT;
    float* hs1 = hs0 + (size_t)BB * TT * HH;

    // Workspace carve-out (256B aligned)
    char* p = (char*)d_ws;
    auto carve = [&](size_t bytes) { char* r = p; p += (bytes + 255) & ~(size_t)255; return r; };
    __bf16* Win0c  = (__bf16*)carve((size_t)DIN * HH * 2);
    __bf16* Wrec0c = (__bf16*)carve((size_t)HH * HH * 2);
    __bf16* Win1c  = (__bf16*)carve((size_t)HH * HH * 2);
    __bf16* Wrec1c = (__bf16*)carve((size_t)HH * HH * 2);
    __bf16* Woutc  = (__bf16*)carve((size_t)HH * DOUT * 2);
    __bf16* P0     = (__bf16*)carve((size_t)TT * BB * HH * 2);
    __bf16* fr0b   = (__bf16*)carve((size_t)2 * BB * HH * 2);
    __bf16* fr1b   = (__bf16*)carve((size_t)2 * BB * HH * 2);
    unsigned* bar  = (unsigned*)carve(256);

    // 1) weight transpose/convert to column-major bf16
    k_cvt_transpose<<<1024, 256, 0, stream>>>(Win0,  Win0c,  DIN, HH);
    k_cvt_transpose<<<2048, 256, 0, stream>>>(Wrec0, Wrec0c, HH,  HH);
    k_cvt_transpose<<<2048, 256, 0, stream>>>(Win1,  Win1c,  HH,  HH);
    k_cvt_transpose<<<2048, 256, 0, stream>>>(Wrec1, Wrec1c, HH,  HH);
    k_cvt_transpose<<<512,  256, 0, stream>>>(Wout,  Woutc,  HH,  DOUT);

    // 2) zero recurrent state buffers + barrier
    k_init<<<(2 * BB * HH + 255) / 256, 256, 0, stream>>>(fr0b, fr1b, bar);

    // 3) input projection P0 = ALPHA*(x@W_in0 + b0)
    k_p0_gemm<<<16384, 128, 0, stream>>>(x, Win0c, b0, P0);

    // 4) persistent recurrence (weights + P0 tile in LDS)
    const size_t ldsBytes = (size_t)3 * COLS_WG * LDSK * sizeof(__bf16)
                          + (size_t)BB * COLS_WG * sizeof(__bf16);   // 200192
    (void)hipFuncSetAttribute((const void*)k_recurrent,
                              hipFuncAttributeMaxDynamicSharedMemorySize, (int)ldsBytes);
    k_recurrent<<<REC_NWG, 128, ldsBytes, stream>>>(P0, Wrec0c, Win1c, Wrec1c, b1,
                                                    hs0, hs1, fr0b, fr1b, bar);

    // 5) readout out = hs1 @ W_out + b_out
    k_readout<<<2048, 128, 0, stream>>>(hs1, Woutc, bout, out);
}
